// GNNModule_84430467105079
// MI455X (gfx1250) — matrix operand order repaired and checked by reading, not compiled, not run on hardware
//
#include <hip/hip_runtime.h>
#include <hip/hip_bf16.h>
#include <stdint.h>

// ---------------------------------------------------------------------------
// GATv2 (2 layers) for MI455X / gfx1250, wave32.
//   layer1: OBS=128 -> 4 heads x 64  (concat -> 256), relu
//   layer2: 256 -> 1 head x 64
// Dense transforms via v_wmma_f32_16x16x32_bf16 with operands PRE-PACKED into
// the wave32 WMMA fragment layouts (ISA 7.12.2), so the GEMM inner loop is
// two 32B contiguous loads + one WMMA. Edge phase via L2-resident gathers +
// encoded-uint atomic segment-max + f32 atomic segment-sum.
// ---------------------------------------------------------------------------

typedef __attribute__((ext_vector_type(16))) __bf16 bf16x16_t;
typedef __attribute__((ext_vector_type(8)))  float  f32x8_t;

#define NEG_SLOPE 0.2f

__device__ __forceinline__ float lrelu(float v) { return v > 0.f ? v : NEG_SLOPE * v; }

// monotone order-preserving float<->uint encoding (for atomic max on floats)
__device__ __forceinline__ unsigned encf(float f) {
  unsigned u = __float_as_uint(f);
  return (u & 0x80000000u) ? ~u : (u | 0x80000000u);
}
__device__ __forceinline__ float decf(unsigned u) {
  return __uint_as_float((u & 0x80000000u) ? (u ^ 0x80000000u) : ~u);
}

// ------------------------------- utility kernels ---------------------------

__global__ void fill_u32_kernel(unsigned* __restrict__ p, unsigned v, long long n) {
  long long i = (long long)blockIdx.x * blockDim.x + threadIdx.x;
  if (i < n) p[i] = v;
}

__global__ void bias_init_kernel(float* __restrict__ out, const float* __restrict__ b,
                                 int cols, long long n) {
  long long i = (long long)blockIdx.x * blockDim.x + threadIdx.x;
  if (i < n) out[i] = b[i % cols];
}

// ------------------------------- operand packing ---------------------------
// A-fragment layout (16-bit A 16x32, wave32):
//   lane l (l16=l&15, half=l>>4) holds row tm*16+l16;
//   elems 0..7  -> k = kb + half*8 + i
//   elems 8..15 -> k = kb + half*8 + 16 + (i-8)
// Packed: Apack[((tm*KB + kb32)*32 + l)*16 + i]   (32 contiguous bytes / lane)
// Optionally fuses bias add + relu (for layer-2 input h) and f32->bf16.

__global__ void pack_a_kernel(const float* __restrict__ in, const float* __restrict__ bias,
                              int relu, __bf16* __restrict__ out, long long M, int K) {
  long long t = (long long)blockIdx.x * blockDim.x + threadIdx.x;
  if (t >= M * K) return;
  const int KB = K >> 5;
  const int i  = (int)(t & 15);
  const int l  = (int)((t >> 4) & 31);
  long long blk = t >> 9;
  const int kb32 = (int)(blk % KB);
  const long long tm = blk / KB;
  const long long row = tm * 16 + (l & 15);
  const int k = kb32 * 32 + (l >> 4) * 8 + (i < 8 ? i : i + 8);
  float v = in[row * K + k];
  if (bias) v += bias[k];
  if (relu) v = v > 0.f ? v : 0.f;
  out[t] = (__bf16)v;
}

// B-fragment layout (16-bit B 32x16, wave32):
//   lane l (half=l>>4): n = tn*16+(l&15)
//   elems 0..7 -> k = kb + half*8 + i; elems 8..15 -> k = kb + half*8 + 16 + (i-8)
// Packed: Bpack[((tn*KB + kb32)*32 + l)*16 + i]

__global__ void pack_b_kernel(const float* __restrict__ B, __bf16* __restrict__ out,
                              int K, int N) {
  long long t = (long long)blockIdx.x * blockDim.x + threadIdx.x;
  if (t >= (long long)K * N) return;
  const int KB = K >> 5;
  const int i  = (int)(t & 15);
  const int l  = (int)((t >> 4) & 31);
  long long blk = t >> 9;
  const int kb32 = (int)(blk % KB);
  const int tn   = (int)(blk / KB);
  const int k = kb32 * 32 + (l >> 4) * 8 + (i < 8 ? i : i + 8);
  const int n = tn * 16 + (l & 15);
  out[t] = (__bf16)B[(long long)k * N + n];
}

// ------------------------------- WMMA GEMM ---------------------------------
// C[M x N] = A * B from pre-packed fragments. One 16x16 tile per wave.

__global__ void gemm_packed_wmma_kernel(const __bf16* __restrict__ Apack,
                                        const __bf16* __restrict__ Bpack,
                                        float* __restrict__ C,
                                        int M, int K, int N) {
  const int lane = threadIdx.x & 31;
  const int wib  = threadIdx.x >> 5;
  const int wpb  = blockDim.x >> 5;
  const int wave = blockIdx.x * wpb + wib;
  const int ntile = N >> 4;
  const int mtile = M >> 4;
  const int KB    = K >> 5;
  if (wave >= ntile * mtile) return;   // whole wave uniform: EXEC stays all-1s
  const int tm = wave / ntile;
  const int tn = wave % ntile;

  const bf16x16_t* ap = (const bf16x16_t*)Apack + (long long)tm * KB * 32 + lane;
  const bf16x16_t* bp = (const bf16x16_t*)Bpack + (long long)tn * KB * 32 + lane;

  f32x8_t acc = {};
#pragma unroll 4
  for (int kb = 0; kb < KB; ++kb) {
    __builtin_prefetch(ap + (long long)(kb + 1) * 32, 0, 0);  // global_prefetch_b8
    bf16x16_t afrag = ap[(long long)kb * 32];
    bf16x16_t bfrag = bp[(long long)kb * 32];
    acc = __builtin_amdgcn_wmma_f32_16x16x32_bf16(
        /*neg_a=*/false, afrag, /*neg_b=*/false, bfrag,
        /*c_mod=*/(short)0, acc, /*reuse_a=*/false, /*reuse_b=*/false);
  }

  // C store: VGPR r, lane<16 -> (M=r,N=lane); lane>=16 -> (M=r+8,N=lane-16)
  const int half = lane >> 4;
  const int l16  = lane & 15;
  float* cp = C + (long long)(tm * 16 + half * 8) * N + tn * 16 + l16;
#pragma unroll
  for (int r = 0; r < 8; ++r) cp[(long long)r * N] = acc[r];
}

// ------------------------------- layer-1 edge phase ------------------------
// 4 heads x 64 dims (256 features). One wave per edge, 8 dims per lane.

__global__ void gat_score1_kernel(const float* __restrict__ XL, const float* __restrict__ XR,
                                  const float* __restrict__ att, const int* __restrict__ ei,
                                  int E_raw, int E_tot,
                                  float* __restrict__ score, unsigned* __restrict__ menc) {
  const int lane = threadIdx.x & 31;
  const long long e = ((long long)blockIdx.x * blockDim.x + threadIdx.x) >> 5;
  if (e >= E_tot) return;
  int src, dst;
  if (e < E_raw) { src = ei[e]; dst = ei[E_raw + e]; }
  else           { src = (int)(e - E_raw); dst = src; }

  const int g = lane * 8;                       // global feature index (head = g/64)
  const float4* pl = (const float4*)(XL + (long long)src * 256 + g);
  const float4* pr = (const float4*)(XR + (long long)dst * 256 + g);
  const float4* pa = (const float4*)(att + g);  // att1 flat [4*64] matches layout
  float s = 0.f;
#pragma unroll
  for (int j = 0; j < 2; ++j) {
    float4 a = pl[j], b = pr[j], w = pa[j];
    s += w.x * lrelu(a.x + b.x);
    s += w.y * lrelu(a.y + b.y);
    s += w.z * lrelu(a.z + b.z);
    s += w.w * lrelu(a.w + b.w);
  }
  // reduce 8 lanes -> one head
  s += __shfl_xor(s, 1, 32);
  s += __shfl_xor(s, 2, 32);
  s += __shfl_xor(s, 4, 32);
  if ((lane & 7) == 0) {
    int h = lane >> 3;
    score[e * 4 + h] = s;
    atomicMax(&menc[(long long)dst * 4 + h], encf(s));
  }
}

__global__ void gat_soft1_kernel(const int* __restrict__ ei, int E_raw, int E_tot,
                                 float* __restrict__ score, const unsigned* __restrict__ menc,
                                 float* __restrict__ denom) {
  long long i = (long long)blockIdx.x * blockDim.x + threadIdx.x;
  if (i >= (long long)E_tot * 4) return;
  long long e = i >> 2;
  int h = (int)(i & 3);
  int dst = (e < E_raw) ? ei[E_raw + e] : (int)(e - E_raw);
  float m  = decf(menc[(long long)dst * 4 + h]);
  float ex = __expf(score[i] - m);
  score[i] = ex;                                   // reuse buffer for exp values
  atomicAdd(&denom[(long long)dst * 4 + h], ex);
}

__global__ void gat_agg1_kernel(const float* __restrict__ XL, const int* __restrict__ ei,
                                int E_raw, int E_tot,
                                const float* __restrict__ ex, const float* __restrict__ denom,
                                float* __restrict__ H) {
  const int lane = threadIdx.x & 31;
  const long long e = ((long long)blockIdx.x * blockDim.x + threadIdx.x) >> 5;
  if (e >= E_tot) return;
  int src, dst;
  if (e < E_raw) { src = ei[e]; dst = ei[E_raw + e]; }
  else           { src = (int)(e - E_raw); dst = src; }

  const int h = lane >> 3;
  const float a = ex[e * 4 + h] / denom[(long long)dst * 4 + h];
  const float4* pl = (const float4*)(XL + (long long)src * 256 + lane * 8);
  float4 v0 = pl[0], v1 = pl[1];
  float* out = H + (long long)dst * 256 + lane * 8;
  atomicAdd(out + 0, v0.x * a);
  atomicAdd(out + 1, v0.y * a);
  atomicAdd(out + 2, v0.z * a);
  atomicAdd(out + 3, v0.w * a);
  atomicAdd(out + 4, v1.x * a);
  atomicAdd(out + 5, v1.y * a);
  atomicAdd(out + 6, v1.z * a);
  atomicAdd(out + 7, v1.w * a);
}

// ------------------------------- layer-2 edge phase ------------------------
// 1 head x 64 dims. One wave per edge, 2 dims per lane.

__global__ void gat_score2_kernel(const float* __restrict__ XL, const float* __restrict__ XR,
                                  const float* __restrict__ att, const int* __restrict__ ei,
                                  int E_raw, int E_tot,
                                  float* __restrict__ score, unsigned* __restrict__ menc) {
  const int lane = threadIdx.x & 31;
  const long long e = ((long long)blockIdx.x * blockDim.x + threadIdx.x) >> 5;
  if (e >= E_tot) return;
  int src, dst;
  if (e < E_raw) { src = ei[e]; dst = ei[E_raw + e]; }
  else           { src = (int)(e - E_raw); dst = src; }

  const int d = lane * 2;
  float2 a = *(const float2*)(XL + (long long)src * 64 + d);
  float2 b = *(const float2*)(XR + (long long)dst * 64 + d);
  float2 w = *(const float2*)(att + d);
  float s = w.x * lrelu(a.x + b.x) + w.y * lrelu(a.y + b.y);
  s += __shfl_xor(s, 1, 32);
  s += __shfl_xor(s, 2, 32);
  s += __shfl_xor(s, 4, 32);
  s += __shfl_xor(s, 8, 32);
  s += __shfl_xor(s, 16, 32);
  if (lane == 0) {
    score[e] = s;
    atomicMax(&menc[dst], encf(s));
  }
}

__global__ void gat_soft2_kernel(const int* __restrict__ ei, int E_raw, int E_tot,
                                 float* __restrict__ score, const unsigned* __restrict__ menc,
                                 float* __restrict__ denom) {
  long long e = (long long)blockIdx.x * blockDim.x + threadIdx.x;
  if (e >= E_tot) return;
  int dst = (e < E_raw) ? ei[E_raw + e] : (int)(e - E_raw);
  float ex = __expf(score[e] - decf(menc[dst]));
  score[e] = ex;
  atomicAdd(&denom[dst], ex);
}

__global__ void gat_agg2_kernel(const float* __restrict__ XL, const int* __restrict__ ei,
                                int E_raw, int E_tot,
                                const float* __restrict__ ex, const float* __restrict__ denom,
                                float* __restrict__ out) {
  const int lane = threadIdx.x & 31;
  const long long e = ((long long)blockIdx.x * blockDim.x + threadIdx.x) >> 5;
  if (e >= E_tot) return;
  int src, dst;
  if (e < E_raw) { src = ei[e]; dst = ei[E_raw + e]; }
  else           { src = (int)(e - E_raw); dst = src; }

  const float a = ex[e] / denom[dst];
  const int d = lane * 2;
  float2 v = *(const float2*)(XL + (long long)src * 64 + d);
  atomicAdd(out + (long long)dst * 64 + d + 0, v.x * a);
  atomicAdd(out + (long long)dst * 64 + d + 1, v.y * a);
}

// ------------------------------- host launcher -----------------------------

extern "C" void kernel_launch(void* const* d_in, const int* in_sizes, int n_in,
                              void* d_out, int out_size, void* d_ws, size_t ws_size,
                              hipStream_t stream) {
  const float* x    = (const float*)d_in[0];
  const int*   ei   = (const int*)  d_in[1];   // [2, E_raw] int32
  const float* W1l  = (const float*)d_in[2];   // [128,256]
  const float* W1r  = (const float*)d_in[3];
  const float* att1 = (const float*)d_in[4];   // [4,64]
  const float* b1   = (const float*)d_in[5];   // [256]
  const float* W2l  = (const float*)d_in[6];   // [256,64]
  const float* W2r  = (const float*)d_in[7];
  const float* att2 = (const float*)d_in[8];   // [1,64]
  const float* b2   = (const float*)d_in[9];   // [64]
  float* out = (float*)d_out;

  const int N     = in_sizes[0] / 128;         // 20000
  const int E_raw = in_sizes[1] / 2;           // 640000
  const int E_tot = E_raw + N;                 // + self loops

  // -------- workspace layout --------
  char* ws = (char*)d_ws;
  size_t off = 0;
  auto alloc = [&](size_t bytes) -> void* {
    void* p = ws + off;
    off = (off + bytes + 255) & ~(size_t)255;
    return p;
  };
  __bf16*   xpk   = (__bf16*)  alloc((size_t)N * 128 * 2);   // packed A, layer 1
  __bf16*   w1lpk = (__bf16*)  alloc((size_t)128 * 256 * 2); // packed B
  __bf16*   w1rpk = (__bf16*)  alloc((size_t)128 * 256 * 2);
  float*    XL1   = (float*)   alloc((size_t)N * 256 * 4);
  float*    XR1   = (float*)   alloc((size_t)N * 256 * 4);
  float*    sc1   = (float*)   alloc((size_t)E_tot * 4 * 4);
  unsigned* m1    = (unsigned*)alloc((size_t)N * 4 * 4);
  float*    dn1   = (float*)   alloc((size_t)N * 4 * 4);
  float*    H     = (float*)   alloc((size_t)N * 256 * 4);
  __bf16*   hpk   = (__bf16*)  alloc((size_t)N * 256 * 2);   // packed A, layer 2
  __bf16*   w2lpk = (__bf16*)  alloc((size_t)256 * 64 * 2);
  __bf16*   w2rpk = (__bf16*)  alloc((size_t)256 * 64 * 2);
  float*    XL2   = (float*)   alloc((size_t)N * 64 * 4);
  float*    XR2   = (float*)   alloc((size_t)N * 64 * 4);
  float*    sc2   = (float*)   alloc((size_t)E_tot * 4);
  unsigned* m2    = (unsigned*)alloc((size_t)N * 4);
  float*    dn2   = (float*)   alloc((size_t)N * 4);
  (void)ws_size; (void)n_in; (void)out_size;

  const int T = 256;
  auto nb  = [](long long n, int t) { return (int)((n + t - 1) / t); };
  auto nbw = [](long long waves)    { return (int)((waves + 7) / 8); };  // 8 waves/block

  // -------- stage 0: pack operands into WMMA fragment layouts --------
  pack_a_kernel<<<nb((long long)N * 128, T), T, 0, stream>>>(x, nullptr, 0, xpk, N, 128);
  pack_b_kernel<<<nb(128 * 256, T), T, 0, stream>>>(W1l, w1lpk, 128, 256);
  pack_b_kernel<<<nb(128 * 256, T), T, 0, stream>>>(W1r, w1rpk, 128, 256);

  // -------- stage 1: node transforms (WMMA) --------
  {
    int tiles = (N / 16) * (256 / 16);
    gemm_packed_wmma_kernel<<<nbw(tiles), T, 0, stream>>>(xpk, w1lpk, XL1, N, 128, 256);
    gemm_packed_wmma_kernel<<<nbw(tiles), T, 0, stream>>>(xpk, w1rpk, XR1, N, 128, 256);
  }

  // -------- stage 2: init accumulators --------
  fill_u32_kernel<<<nb((long long)N * 256, T), T, 0, stream>>>((unsigned*)H, 0u, (long long)N * 256);
  fill_u32_kernel<<<nb((long long)N * 4, T), T, 0, stream>>>(m1, 0u, (long long)N * 4);
  fill_u32_kernel<<<nb((long long)N * 4, T), T, 0, stream>>>((unsigned*)dn1, 0u, (long long)N * 4);
  fill_u32_kernel<<<nb(N, T), T, 0, stream>>>(m2, 0u, N);
  fill_u32_kernel<<<nb(N, T), T, 0, stream>>>((unsigned*)dn2, 0u, N);
  bias_init_kernel<<<nb((long long)N * 64, T), T, 0, stream>>>(out, b2, 64, (long long)N * 64);

  // -------- stage 3: layer-1 edge phase --------
  gat_score1_kernel<<<nbw(E_tot), T, 0, stream>>>(XL1, XR1, att1, ei, E_raw, E_tot, sc1, m1);
  gat_soft1_kernel<<<nb((long long)E_tot * 4, T), T, 0, stream>>>(ei, E_raw, E_tot, sc1, m1, dn1);
  gat_agg1_kernel<<<nbw(E_tot), T, 0, stream>>>(XL1, ei, E_raw, E_tot, sc1, dn1, H);

  // -------- stage 4: h = relu(H + b1) packed to bf16 fragments; layer-2 GEMMs ----
  pack_a_kernel<<<nb((long long)N * 256, T), T, 0, stream>>>(H, b1, 1, hpk, N, 256);
  pack_b_kernel<<<nb(256 * 64, T), T, 0, stream>>>(W2l, w2lpk, 256, 64);
  pack_b_kernel<<<nb(256 * 64, T), T, 0, stream>>>(W2r, w2rpk, 256, 64);
  {
    int tiles = (N / 16) * (64 / 16);
    gemm_packed_wmma_kernel<<<nbw(tiles), T, 0, stream>>>(hpk, w2lpk, XL2, N, 256, 64);
    gemm_packed_wmma_kernel<<<nbw(tiles), T, 0, stream>>>(hpk, w2rpk, XR2, N, 256, 64);
  }

  // -------- stage 5: layer-2 edge phase (accumulates into bias-initialized out) --------
  gat_score2_kernel<<<nbw(E_tot), T, 0, stream>>>(XL2, XR2, att2, ei, E_raw, E_tot, sc2, m2);
  gat_soft2_kernel<<<nb(E_tot, T), T, 0, stream>>>(ei, E_raw, E_tot, sc2, m2, dn2);
  gat_agg2_kernel<<<nbw(E_tot), T, 0, stream>>>(XL2, ei, E_raw, E_tot, sc2, dn2, out);
}
